// DioLstm_51058571215539
// MI455X (gfx1250) — matrix operand order repaired and checked by reading, not compiled, not run on hardware
//
#include <hip/hip_runtime.h>

// Problem constants (from reference): B=4, n=24 words, H=512.
#define BN   4
#define NW   24
#define HH   512
#define H2   1024           // 2H
#define H5   2560           // 5H

typedef __attribute__((ext_vector_type(16))) __bf16 v16bf;
typedef __attribute__((ext_vector_type(8)))  float  v8f;

union ABfrag { v16bf v; uint4 q[2]; unsigned int u[8]; unsigned short s[16]; };

// ---- helpers -------------------------------------------------------------
__device__ __forceinline__ unsigned short f2bf(float x) {
    unsigned int u = __float_as_uint(x);
    u += 0x7FFFu + ((u >> 16) & 1u);      // round-to-nearest-even
    return (unsigned short)(u >> 16);
}
__device__ __forceinline__ float sigm(float x) { return 1.0f / (1.0f + __expf(-x)); }

// ---- prep kernels --------------------------------------------------------
__global__ void k_cvt_bf16(const float* __restrict__ src, unsigned short* __restrict__ dst, int count) {
    int i = blockIdx.x * blockDim.x + threadIdx.x;
    if (i < count) dst[i] = f2bf(src[i]);
}

// Wbt[nn*2H + k] = bf16(Wbil[k*2H + nn])   (transpose so K is contiguous per column)
__global__ void k_cvt_wbil_t(const float* __restrict__ src, unsigned short* __restrict__ dst) {
    int i = blockIdx.x * blockDim.x + threadIdx.x;
    if (i < H2 * H2) {
        int nn = i / H2, k = i % H2;
        dst[i] = f2bf(src[k * H2 + nn]);
    }
}

// bias_total = bi + bs + ins_bias (ones on [H,3H))
__global__ void k_bias(const float* __restrict__ bi, const float* __restrict__ bs,
                       float* __restrict__ bt) {
    int j = blockIdx.x * blockDim.x + threadIdx.x;
    if (j < H5) {
        float ib = (j >= HH && j < 3 * HH) ? 1.0f : 0.0f;
        bt[j] = bi[j] + bs[j] + ib;
    }
}

__global__ void k_zero_f32(float* __restrict__ p, int count) {
    int i = blockIdx.x * blockDim.x + threadIdx.x;
    if (i < count) p[i] = 0.0f;
}

__global__ void k_zero_u16(unsigned short* __restrict__ p, int count) {
    int i = blockIdx.x * blockDim.x + threadIdx.x;
    if (i < count) p[i] = 0;
}

// leaf cells (len=0): chartH/C from seqt, plus bf16 [h,c] row
__global__ void k_leaf(const float* __restrict__ seqt, float* __restrict__ chartH,
                       float* __restrict__ chartC, unsigned short* __restrict__ chartVb) {
    int i = blockIdx.x * blockDim.x + threadIdx.x;
    if (i >= BN * NW * HH) return;
    int b = i / (NW * HH);
    int r = i % (NW * HH);
    int left = r / HH, j = r % HH;
    int cell = (b * NW + left) * NW + 0;
    float h = seqt[(size_t)(b * NW + left) * H2 + j];
    float c = seqt[(size_t)(b * NW + left) * H2 + HH + j];
    chartH[(size_t)cell * HH + j] = h;
    chartC[(size_t)cell * HH + j] = c;
    chartVb[(size_t)cell * H2 + j]      = f2bf(h);
    chartVb[(size_t)cell * H2 + HH + j] = f2bf(c);
}

// ---- per-diagonal combine: scores, softmax over splits, gates, update ----
__global__ __launch_bounds__(256)
void k_combine(float* __restrict__ chartH, float* __restrict__ chartC,
               float* __restrict__ chartS, unsigned short* __restrict__ chartVb,
               const float* __restrict__ Pi, const float* __restrict__ Ps,
               const float* __restrict__ tArr, const float* __restrict__ biasT,
               int L, int P) {
    int b = blockIdx.x / P;
    int p = blockIdx.x % P;
    int tid = threadIdx.x;

    __shared__ float sred[256];
    __shared__ float scompat[32];

    // pass 1: compat[k] = t[left,k] . [rH,rC] + lS + rS
    for (int k = 0; k < L; ++k) {
        int lc = (b * NW + p) * NW + k;
        int rc = (b * NW + (p + k + 1)) * NW + (L - 1 - k);
        const float* tv = tArr + (size_t)lc * H2;
        float part = 0.0f;
        for (int j = tid; j < H2; j += 256) {
            float rv = (j < HH) ? chartH[(size_t)rc * HH + j]
                                : chartC[(size_t)rc * HH + (j - HH)];
            part += tv[j] * rv;
        }
        sred[tid] = part; __syncthreads();
        for (int s = 128; s > 0; s >>= 1) {
            if (tid < s) sred[tid] += sred[tid + s];
            __syncthreads();
        }
        if (tid == 0) scompat[k] = sred[0] + chartS[lc] + chartS[rc];
        __syncthreads();
    }

    // softmax over k (redundant per thread; L <= 23)
    float cmax = -3.4e38f;
    for (int k = 0; k < L; ++k) cmax = fmaxf(cmax, scompat[k]);
    float w[NW]; float den = 0.0f;
    for (int k = 0; k < L; ++k) { w[k] = __expf(scompat[k] - cmax); den += w[k]; }
    float inv = 1.0f / den;
    float Snew = 0.0f;
    for (int k = 0; k < L; ++k) { w[k] *= inv; Snew += w[k] * scompat[k]; }

    // pass 2: gates + weighted accumulation (each thread owns 2 of H=512 lanes)
    float accH[2] = {0.0f, 0.0f}, accC[2] = {0.0f, 0.0f};
    for (int k = 0; k < L; ++k) {
        int lc = (b * NW + p) * NW + k;
        int rc = (b * NW + (p + k + 1)) * NW + (L - 1 - k);
        const float* pi = Pi + (size_t)lc * H5;
        const float* ps = Ps + (size_t)rc * H5;
        float wk = w[k];
        #pragma unroll
        for (int q = 0; q < 2; ++q) {
            int j = tid + q * 256;
            float ig = sigm (pi[j]          + ps[j]          + biasT[j]);
            float lf = sigm (pi[HH + j]     + ps[HH + j]     + biasT[HH + j]);
            float rf = sigm (pi[2*HH + j]   + ps[2*HH + j]   + biasT[2*HH + j]);
            float g  = tanhf(pi[3*HH + j]   + ps[3*HH + j]   + biasT[3*HH + j]);
            float og = sigm (pi[4*HH + j]   + ps[4*HH + j]   + biasT[4*HH + j]);
            float lCv = chartC[(size_t)lc * HH + j];
            float rCv = chartC[(size_t)rc * HH + j];
            float mem = lf * lCv + rf * rCv + ig * g;
            float h = og * tanhf(mem);
            accH[q] += wk * h;
            accC[q] += wk * mem;
        }
    }

    int nc = (b * NW + p) * NW + L;
    #pragma unroll
    for (int q = 0; q < 2; ++q) {
        int j = tid + q * 256;
        chartH[(size_t)nc * HH + j] = accH[q];
        chartC[(size_t)nc * HH + j] = accC[q];
        chartVb[(size_t)nc * H2 + j]      = f2bf(accH[q]);
        chartVb[(size_t)nc * H2 + HH + j] = f2bf(accC[q]);
    }
    if (tid == 0) chartS[nc] = Snew;
}

// ---- WMMA bf16 GEMM: Out[cell(m), ncol] = sum_k A[cell(m), k] * W[ncol, k] ----
// A rows are bf16 chart rows (chartVb, stride 2H); W is [N, K] bf16, K contiguous.
// One wave computes one 16x16 tile; blockDim=32 => EXEC all ones for WMMA.
// Padded rows (m >= Mrows) read from a zeroed dummy row: no divergent loads.
// Fragment elements are two K-contiguous runs of 8 bf16 => two b128 loads each.
template <int KD>
__global__ __launch_bounds__(32)
void k_gemm_bf16(const unsigned short* __restrict__ Avb,
                 const unsigned short* __restrict__ zrow,
                 const unsigned short* __restrict__ W,
                 float* __restrict__ Out, int Mrows, int Pdiag, int L, int outStride) {
    int lane  = threadIdx.x;
    int mtile = blockIdx.x;
    int ntile = blockIdx.y;
    int mloc  = lane & 15;
    int khalf = lane >> 4;

    int m = mtile * 16 + mloc;
    const unsigned short* arowp;
    if (m < Mrows) {
        int b = m / Pdiag, p = m % Pdiag;
        arowp = Avb + (size_t)((b * NW + p) * NW + L) * H2;
    } else {
        arowp = zrow;                     // zero-padded dummy row
    }
    int ncol = ntile * 16 + mloc;
    const unsigned short* wp = W + (size_t)ncol * KD;

    v8f acc = {};
    #pragma unroll
    for (int kb = 0; kb < KD; kb += 32) {
        ABfrag a, bm;
        // A 16x32 bf16: lane half khalf holds K runs [khalf*8, +8) and [16+khalf*8, +8)
        a.q[0] = *(const uint4*)(arowp + kb + khalf * 8);
        a.q[1] = *(const uint4*)(arowp + kb + 16 + khalf * 8);
        // B 32x16 bf16: lane half khalf holds K runs [khalf*8, +8) and [16+khalf*8, +8), col = lane&15
        bm.q[0] = *(const uint4*)(wp + kb + khalf * 8);
        bm.q[1] = *(const uint4*)(wp + kb + 16 + khalf * 8);
        acc = __builtin_amdgcn_wmma_f32_16x16x32_bf16(
            false, a.v, false, bm.v, (short)0, acc, false, false);
    }

    #pragma unroll
    for (int r = 0; r < 8; ++r) {
        int mr = mtile * 16 + khalf * 8 + r;   // D: VGPR r -> M=r (lanes 0-15) / M=r+8 (lanes 16-31)
        if (mr < Mrows) {
            int b = mr / Pdiag, p = mr % Pdiag;
            size_t cell = (size_t)((b * NW + p) * NW + L);
            Out[cell * outStride + ncol] = acc[r];
        }
    }
}

// ---- final root extraction: out[b] = concat(h, c) of cell (0, n-1) -------
__global__ void k_root(const float* __restrict__ chartH, const float* __restrict__ chartC,
                       float* __restrict__ out) {
    int i = blockIdx.x * blockDim.x + threadIdx.x;
    if (i >= BN * H2) return;
    int b = i / H2, j = i % H2;
    int cell = (b * NW + 0) * NW + (NW - 1);
    out[i] = (j < HH) ? chartH[(size_t)cell * HH + j]
                      : chartC[(size_t)cell * HH + (j - HH)];
}

// ---- host ----------------------------------------------------------------
extern "C" void kernel_launch(void* const* d_in, const int* in_sizes, int n_in,
                              void* d_out, int out_size, void* d_ws, size_t ws_size,
                              hipStream_t stream) {
    (void)in_sizes; (void)n_in; (void)out_size; (void)ws_size;
    const float* seqt = (const float*)d_in[0];
    const float* Wi   = (const float*)d_in[1];
    const float* bi   = (const float*)d_in[2];
    const float* Ws   = (const float*)d_in[3];
    const float* bs   = (const float*)d_in[4];
    const float* Wbil = (const float*)d_in[5];
    float* out = (float*)d_out;

    char* ws = (char*)d_ws;
    auto alloc = [&](size_t bytes) -> void* {
        void* p = (void*)ws;
        ws += (bytes + 255) & ~(size_t)255;
        return p;
    };
    const int nCells = BN * NW * NW;
    unsigned short* Wib     = (unsigned short*)alloc((size_t)H5 * HH * 2);
    unsigned short* Wsb     = (unsigned short*)alloc((size_t)H5 * HH * 2);
    unsigned short* Wbt     = (unsigned short*)alloc((size_t)H2 * H2 * 2);
    float*          biasT   = (float*)alloc((size_t)H5 * 4);
    float*          chartH  = (float*)alloc((size_t)nCells * HH * 4);
    float*          chartC  = (float*)alloc((size_t)nCells * HH * 4);
    float*          chartS  = (float*)alloc((size_t)nCells * 4);
    unsigned short* chartVb = (unsigned short*)alloc((size_t)nCells * H2 * 2);
    unsigned short* zrow    = (unsigned short*)alloc((size_t)H2 * 2);
    float*          Pi      = (float*)alloc((size_t)nCells * H5 * 4);
    float*          Ps      = (float*)alloc((size_t)nCells * H5 * 4);
    float*          tArr    = (float*)alloc((size_t)nCells * H2 * 4);

    // one-time prep (per call; deterministic)
    k_cvt_bf16 <<<(H5 * HH + 255) / 256, 256, 0, stream>>>(Wi, Wib, H5 * HH);
    k_cvt_bf16 <<<(H5 * HH + 255) / 256, 256, 0, stream>>>(Ws, Wsb, H5 * HH);
    k_cvt_wbil_t<<<(H2 * H2 + 255) / 256, 256, 0, stream>>>(Wbil, Wbt);
    k_bias     <<<(H5 + 255) / 256, 256, 0, stream>>>(bi, bs, biasT);
    k_zero_f32 <<<(nCells + 255) / 256, 256, 0, stream>>>(chartS, nCells);
    k_zero_u16 <<<(H2 + 255) / 256, 256, 0, stream>>>(zrow, H2);
    k_leaf     <<<(BN * NW * HH + 255) / 256, 256, 0, stream>>>(seqt, chartH, chartC, chartVb);

    auto proj = [&](int Mrows, int Pdiag, int L) {
        dim3 g5((unsigned)((Mrows + 15) / 16), H5 / 16);
        k_gemm_bf16<HH><<<g5, 32, 0, stream>>>(chartVb, zrow, Wib, Pi,   Mrows, Pdiag, L, H5);
        k_gemm_bf16<HH><<<g5, 32, 0, stream>>>(chartVb, zrow, Wsb, Ps,   Mrows, Pdiag, L, H5);
        dim3 g2((unsigned)((Mrows + 15) / 16), H2 / 16);
        k_gemm_bf16<H2><<<g2, 32, 0, stream>>>(chartVb, zrow, Wbt, tArr, Mrows, Pdiag, L, H2);
    };

    // leaf projections (diagonal L=0)
    proj(BN * NW, NW, 0);

    // inside pass over diagonals
    for (int L = 1; L < NW; ++L) {
        int P = NW - L;
        k_combine<<<BN * P, 256, 0, stream>>>(chartH, chartC, chartS, chartVb,
                                              Pi, Ps, tArr, biasT, L, P);
        if (L < NW - 1) proj(BN * P, P, L);   // root projections never consumed
    }

    k_root<<<(BN * H2 + 255) / 256, 256, 0, stream>>>(chartH, chartC, out);
}
